// RPNLayer_36885179138100
// MI455X (gfx1250) — compile-verified, matrix operand order checked by reading, not run on hardware
//
#include <hip/hip_runtime.h>
#include <math.h>

// ---------------- types ----------------
typedef __bf16 bf16_t;
typedef __attribute__((ext_vector_type(16))) __bf16 v16bf;
typedef __attribute__((ext_vector_type(8)))  float  v8f;
typedef __attribute__((ext_vector_type(4)))  unsigned int u32x4;

union FragU { u32x4 q[2]; v16bf v; };

// ---------------- problem dims ----------------
static constexpr int H  = 100;
static constexpr int W  = 152;
static constexpr int CIN = 1024;
static constexpr int CF  = 512;
static constexpr int P   = H * W;          // 15200 pixels
static constexpr int NA  = P * 9;          // 136800 anchors
static constexpr int WP  = W + 2;          // padded width
static constexpr int HP  = H + 2;
static constexpr int PRE_N  = 6000;
static constexpr int POST_N = 300;
static constexpr int SORT_N = 8192;
static constexpr int NBINS  = 4096;

// d_out layout (floats): bbox_pred | cls_scores | rois | anchors
static constexpr size_t BBOX_OFF = 0;
static constexpr size_t CLS_OFF  = (size_t)NA * 4;             // 547200
static constexpr size_t ROIS_OFF = CLS_OFF + (size_t)NA * 2;   // 820800
static constexpr size_t ANCH_OFF = ROIS_OFF + (size_t)POST_N*4;// 822000

// ---------------- K1: pad + f32->bf16 NHWC repack of input ----------------
__global__ __launch_bounds__(256) void k_pad_input(const float* __restrict__ x,
                                                   bf16_t* __restrict__ xp) {
  int i = blockIdx.x * 256 + threadIdx.x;           // over HP*WP*CIN = 16,084,992
  int c   = i & (CIN - 1);
  int pix = i >> 10;
  int wp = pix % WP, hp = pix / WP;
  float v = 0.f;
  if (hp >= 1 && hp <= H && wp >= 1 && wp <= W)
    v = x[(size_t)c * P + (size_t)(hp - 1) * W + (wp - 1)];
  xp[i] = (bf16_t)v;
}

// ---------------- K2: conv3x3 weights -> [kk][n][c] bf16 ----------------
__global__ __launch_bounds__(256) void k_prep_w(const float* __restrict__ w,
                                                bf16_t* __restrict__ wT) {
  int i = blockIdx.x * 256 + threadIdx.x;           // over 9*512*1024 = 4,718,592
  int c = i & (CIN - 1);
  int t = i >> 10;
  int n = t % CF;
  int kk = t / CF;
  wT[i] = (bf16_t)w[((size_t)n * CIN + c) * 9 + kk];
}

// ---------------- K2b: combined padded 1x1 weights [80][512] bf16 ----------
__global__ __launch_bounds__(256) void k_prep_wc(const float* __restrict__ cw,
                                                 const float* __restrict__ rw,
                                                 bf16_t* __restrict__ wc) {
  int i = blockIdx.x * 256 + threadIdx.x;           // over 80*512 = 40960
  if (i >= 80 * CF) return;
  int c = i & (CF - 1);
  int n = i >> 9;
  float v = 0.f;
  if (n < 32) { if (n < 18) v = cw[(size_t)n * CF + c]; }
  else        { int m = n - 32; if (m < 36) v = rw[(size_t)m * CF + c]; }
  wc[i] = (bf16_t)v;
}

// ---------------- K3: 3x3 conv as implicit GEMM via WMMA bf16 --------------
// block = 8 waves; block tile = 32 pixels x 512 channels; wave = 32px x 64ch.
__global__ __launch_bounds__(256) void k_conv3x3(const bf16_t* __restrict__ xp,
                                                 const bf16_t* __restrict__ wT,
                                                 const float* __restrict__ bias,
                                                 bf16_t* __restrict__ fm) {
  const int lane = threadIdx.x & 31;
  const int wv   = threadIdx.x >> 5;        // 0..7
  const int nw   = wv * 64;                 // wave's channel base
  const int l15  = lane & 15;
  const int khi  = (lane >> 4) << 3;        // 0 or 8 : K-offset of fragment
  const int mh   = khi;                     // 0 or 8 : M-offset of D rows
  const int pix0 = blockIdx.x * 32;

  const int p0 = pix0 + l15, p1 = pix0 + 16 + l15;
  const size_t base0 = (size_t)(p0 / W) * WP + (p0 % W);
  const size_t base1 = (size_t)(p1 / W) * WP + (p1 % W);

  v8f acc[2][4] = {};

  #pragma unroll 1
  for (int kk = 0; kk < 9; ++kk) {
    const int kh = kk / 3, kw = kk % 3;
    const bf16_t* a0 = xp + (((base0 + (size_t)kh * WP + kw) << 10) + khi);
    const bf16_t* a1 = xp + (((base1 + (size_t)kh * WP + kw) << 10) + khi);
    const bf16_t* bp[4];
    #pragma unroll
    for (int bj = 0; bj < 4; ++bj)
      bp[bj] = wT + (((size_t)kk * CF + nw + bj * 16 + l15) << 10) + khi;

    #pragma unroll 2
    for (int c0 = 0; c0 < CIN; c0 += 32) {
      FragU fa0, fa1;
      fa0.q[0] = *(const u32x4*)(a0 + c0);
      fa0.q[1] = *(const u32x4*)(a0 + c0 + 16);
      fa1.q[0] = *(const u32x4*)(a1 + c0);
      fa1.q[1] = *(const u32x4*)(a1 + c0 + 16);
      #pragma unroll
      for (int bj = 0; bj < 4; ++bj) {
        FragU fb;
        fb.q[0] = *(const u32x4*)(bp[bj] + c0);
        fb.q[1] = *(const u32x4*)(bp[bj] + c0 + 16);
        acc[0][bj] = __builtin_amdgcn_wmma_f32_16x16x32_bf16(
            false, fa0.v, false, fb.v, (short)0, acc[0][bj], false, false);
        acc[1][bj] = __builtin_amdgcn_wmma_f32_16x16x32_bf16(
            false, fa1.v, false, fb.v, (short)0, acc[1][bj], false, false);
      }
    }
  }

  // epilogue: bias + ReLU, store bf16 NHWC fm[pixel][channel]
  #pragma unroll
  for (int bj = 0; bj < 4; ++bj) {
    const int n = nw + bj * 16 + l15;
    const float b = bias[n];
    #pragma unroll
    for (int mi = 0; mi < 2; ++mi) {
      const int pbase = pix0 + mi * 16 + mh;
      #pragma unroll
      for (int r = 0; r < 8; ++r) {
        float v = acc[mi][bj][r] + b;
        v = v > 0.f ? v : 0.f;
        fm[(size_t)(pbase + r) * CF + n] = (bf16_t)v;
      }
    }
  }
}

// ---------------- K4: 1x1 convs (cls+reg) via WMMA, transposed scatter -----
__global__ __launch_bounds__(256) void k_conv1x1(const bf16_t* __restrict__ fm,
                                                 const bf16_t* __restrict__ wc,
                                                 const float* __restrict__ clsb,
                                                 const float* __restrict__ regb,
                                                 float* __restrict__ out) {
  const int lane = threadIdx.x & 31;
  const int wv   = threadIdx.x >> 5;
  const int job  = blockIdx.x * 8 + wv;     // 950 m-tiles x 5 n-tiles
  if (job >= 950 * 5) return;
  const int mt = job / 5, nt = job % 5;
  const int l15 = lane & 15;
  const int khi = (lane >> 4) << 3;
  const int mh  = khi;

  const bf16_t* ap = fm + ((size_t)(mt * 16 + l15) << 9) + khi;
  const bf16_t* bp = wc + ((size_t)(nt * 16 + l15) << 9) + khi;

  v8f acc = {};
  #pragma unroll 4
  for (int c0 = 0; c0 < CF; c0 += 32) {
    FragU fa, fb;
    fa.q[0] = *(const u32x4*)(ap + c0);
    fa.q[1] = *(const u32x4*)(ap + c0 + 16);
    fb.q[0] = *(const u32x4*)(bp + c0);
    fb.q[1] = *(const u32x4*)(bp + c0 + 16);
    acc = __builtin_amdgcn_wmma_f32_16x16x32_bf16(
        false, fa.v, false, fb.v, (short)0, acc, false, false);
  }

  const int ch = nt * 16 + l15;             // combined channel 0..79
  #pragma unroll
  for (int r = 0; r < 8; ++r) {
    const int pix = mt * 16 + mh + r;
    float v = acc[r];
    if (ch < 32) {                          // cls path (18 real channels)
      if (ch < 18) {
        v += clsb[ch];
        out[CLS_OFF + ((size_t)pix * 9 + (ch >> 1)) * 2 + (ch & 1)] = v;
      }
    } else {                                // reg path (36 real channels)
      const int c = ch - 32;
      if (c < 36) {
        v += regb[c];
        out[BBOX_OFF + ((size_t)pix * 9 + (c >> 2)) * 4 + (c & 3)] = v;
      }
    }
  }
}

// ---------------- K5: anchors + decode + clip + objectness -----------------
__global__ __launch_bounds__(256) void k_decode(float* __restrict__ out,
                                                float* __restrict__ boxes,
                                                float* __restrict__ scores,
                                                const int* __restrict__ imh_p,
                                                const int* __restrict__ imw_p) {
  int i = blockIdx.x * 256 + threadIdx.x;
  if (i >= NA) return;
  const int pix = i / 9, a = i % 9;
  const int y = pix / W, x = pix % W;
  const int rr = a / 3, ss = a % 3;
  const float scl[3] = {8.f, 16.f, 32.f};
  const float rat[3] = {0.5f, 1.f, 2.f};
  const float sq = sqrtf(rat[rr]);
  const float ws = 16.f * scl[ss] / sq;
  const float hs = 16.f * scl[ss] * sq;
  const float acx = (x + 0.5f) * 16.f, acy = (y + 0.5f) * 16.f;

  out[ANCH_OFF + (size_t)i * 4 + 0] = acx - 0.5f * ws;
  out[ANCH_OFF + (size_t)i * 4 + 1] = acy - 0.5f * hs;
  out[ANCH_OFF + (size_t)i * 4 + 2] = acx + 0.5f * ws;
  out[ANCH_OFF + (size_t)i * 4 + 3] = acy + 0.5f * hs;

  const float dx = out[BBOX_OFF + (size_t)i * 4 + 0];
  const float dy = out[BBOX_OFF + (size_t)i * 4 + 1];
  const float dw = fminf(out[BBOX_OFF + (size_t)i * 4 + 2], 4.1352f);
  const float dh = fminf(out[BBOX_OFF + (size_t)i * 4 + 3], 4.1352f);
  const float pcx = dx * ws + acx, pcy = dy * hs + acy;
  const float pw = expf(dw) * ws, ph = expf(dh) * hs;

  const float imw = (float)imw_p[0], imh = (float)imh_p[0];
  const float b0 = fminf(fmaxf(pcx - 0.5f * pw, 0.f), imw);
  const float b1 = fminf(fmaxf(pcy - 0.5f * ph, 0.f), imh);
  const float b2 = fminf(fmaxf(pcx + 0.5f * pw, 0.f), imw);
  const float b3 = fminf(fmaxf(pcy + 0.5f * ph, 0.f), imh);
  boxes[(size_t)i * 4 + 0] = b0;
  boxes[(size_t)i * 4 + 1] = b1;
  boxes[(size_t)i * 4 + 2] = b2;
  boxes[(size_t)i * 4 + 3] = b3;

  const bool valid = (b2 - b0 >= 16.f) && (b3 - b1 >= 16.f);
  const float s0 = out[CLS_OFF + (size_t)i * 2 + 0];
  const float s1 = out[CLS_OFF + (size_t)i * 2 + 1];
  const float obj = 1.f / (1.f + expf(s0 - s1));       // softmax[:,1]
  scores[i] = valid ? obj : -INFINITY;
}

// ---------------- K6: init histogram / keys / counters ---------------------
__global__ __launch_bounds__(256) void k_init(unsigned int* __restrict__ hist,
                                              unsigned long long* __restrict__ keys,
                                              int* __restrict__ tinfo) {
  int i = blockIdx.x * 256 + threadIdx.x;   // 8192 threads
  if (i < NBINS) hist[i] = 0u;
  if (i < SORT_N) keys[i] = ~0ULL;          // sentinel sorts last
  if (i < 8) tinfo[i] = 0;
}

// ---------------- K7: score histogram --------------------------------------
__global__ __launch_bounds__(256) void k_hist(const float* __restrict__ scores,
                                              unsigned int* __restrict__ hist) {
  int i = blockIdx.x * 256 + threadIdx.x;
  if (i >= NA) return;
  const float s = scores[i];
  if (s >= 0.f) {
    int b = (int)(s * (float)NBINS);
    b = b > NBINS - 1 ? NBINS - 1 : b;
    atomicAdd(&hist[b], 1u);
  }
}

// ---------------- K8: find top-6000 threshold bin --------------------------
__global__ void k_thresh(const unsigned int* __restrict__ hist,
                         int* __restrict__ tinfo) {
  if (threadIdx.x != 0) return;
  int total = 0, T = -1, cntHi = 0;
  for (int b = NBINS - 1; b >= 0; --b) {
    if (total + (int)hist[b] >= PRE_N) { T = b; cntHi = total; break; }
    total += (int)hist[b];
  }
  if (T < 0) cntHi = total;                 // fewer than PRE_N valid scores
  tinfo[0] = T;
  tinfo[1] = cntHi;
  tinfo[2] = (T >= 0) ? (PRE_N - cntHi) : 0;
  tinfo[3] = 0;                             // counter: bins > T
  tinfo[4] = 0;                             // counter: bin == T
}

// ---------------- K9: compact top candidates into key array ----------------
__global__ __launch_bounds__(256) void k_compact(const float* __restrict__ scores,
                                                 unsigned long long* __restrict__ keys,
                                                 int* __restrict__ tinfo) {
  int i = blockIdx.x * 256 + threadIdx.x;
  if (i >= NA) return;
  const float s = scores[i];
  if (s < 0.f) return;
  int b = (int)(s * (float)NBINS);
  b = b > NBINS - 1 ? NBINS - 1 : b;
  const int T = tinfo[0];
  unsigned int mb = __float_as_uint(s);
  mb = (mb & 0x80000000u) ? ~mb : (mb | 0x80000000u);   // monotonic map
  const unsigned long long key =
      ((unsigned long long)(~mb) << 32) | (unsigned int)i; // asc = score desc, idx asc
  if (b > T) {
    const int pos = atomicAdd(&tinfo[3], 1);
    keys[pos] = key;
  } else if (b == T) {
    const int pos = atomicAdd(&tinfo[4], 1);
    if (pos < tinfo[2]) keys[tinfo[1] + pos] = key;
  }
}

// ---------------- K10: single-block bitonic sort of 8192 keys --------------
__global__ __launch_bounds__(1024) void k_sort(unsigned long long* __restrict__ keys) {
  extern __shared__ unsigned long long lds[];
  const int tid = threadIdx.x;
  for (int i = tid; i < SORT_N; i += 1024) lds[i] = keys[i];
  for (int size = 2; size <= SORT_N; size <<= 1) {
    for (int stride = size >> 1; stride > 0; stride >>= 1) {
      __syncthreads();
      for (int t = tid; t < SORT_N; t += 1024) {
        const int p = t ^ stride;
        if (p > t) {
          const unsigned long long a = lds[t], b = lds[p];
          const bool asc = (t & size) == 0;
          if ((a > b) == asc) { lds[t] = b; lds[p] = a; }
        }
      }
    }
  }
  __syncthreads();
  for (int i = tid; i < SORT_N; i += 1024) keys[i] = lds[i];
}

// ---------------- K11: gather top-6000 boxes in score order ----------------
__global__ __launch_bounds__(256) void k_gather(const unsigned long long* __restrict__ keys,
                                                const float* __restrict__ boxes,
                                                float* __restrict__ topBoxes) {
  int r = blockIdx.x * 256 + threadIdx.x;
  if (r >= PRE_N) return;
  const unsigned long long k = keys[r];
  float b0 = 0.f, b1 = 0.f, b2 = 0.f, b3 = 0.f;
  if (k != ~0ULL) {
    const unsigned int idx = (unsigned int)k;
    b0 = boxes[(size_t)idx * 4 + 0];
    b1 = boxes[(size_t)idx * 4 + 1];
    b2 = boxes[(size_t)idx * 4 + 2];
    b3 = boxes[(size_t)idx * 4 + 3];
  }
  topBoxes[(size_t)r * 4 + 0] = b0;
  topBoxes[(size_t)r * 4 + 1] = b1;
  topBoxes[(size_t)r * 4 + 2] = b2;
  topBoxes[(size_t)r * 4 + 3] = b3;
}

// ---------------- K12: NMS (sequential i, parallel j) + write rois ---------
__global__ __launch_bounds__(1024) void k_nms(const float* __restrict__ topBoxes,
                                              float* __restrict__ rois) {
  extern __shared__ float bx[];             // PRE_N * 4 floats (96 KB)
  __shared__ unsigned char keep[PRE_N];
  __shared__ int kidx[POST_N];
  __shared__ int kcnt;
  const int tid = threadIdx.x;
  for (int i = tid; i < PRE_N * 4; i += 1024) bx[i] = topBoxes[i];
  for (int i = tid; i < PRE_N; i += 1024) keep[i] = 1;

  for (int i = 0; i < PRE_N; ++i) {
    __syncthreads();
    if (!keep[i]) continue;                 // uniform across block
    const float x1 = bx[i * 4 + 0], y1 = bx[i * 4 + 1];
    const float x2 = bx[i * 4 + 2], y2 = bx[i * 4 + 3];
    const float ai = (x2 - x1) * (y2 - y1);
    for (int j = i + 1 + tid; j < PRE_N; j += 1024) {
      if (!keep[j]) continue;
      const float xx1 = fmaxf(x1, bx[j * 4 + 0]);
      const float yy1 = fmaxf(y1, bx[j * 4 + 1]);
      const float xx2 = fminf(x2, bx[j * 4 + 2]);
      const float yy2 = fminf(y2, bx[j * 4 + 3]);
      const float inter = fmaxf(xx2 - xx1, 0.f) * fmaxf(yy2 - yy1, 0.f);
      const float aj = (bx[j * 4 + 2] - bx[j * 4 + 0]) * (bx[j * 4 + 3] - bx[j * 4 + 1]);
      const float iou = inter / (ai + aj - inter + 1e-9f);
      if (iou > 0.7f) keep[j] = 0;
    }
  }
  __syncthreads();
  if (tid == 0) {
    int c = 0;
    for (int i = 0; i < PRE_N && c < POST_N; ++i)
      if (keep[i]) kidx[c++] = i;
    kcnt = c;
  }
  __syncthreads();
  if (tid < POST_N) {
    float o0 = 0.f, o1 = 0.f, o2 = 0.f, o3 = 0.f;
    if (tid < kcnt) {
      const int s = kidx[tid];
      o0 = bx[s * 4 + 0]; o1 = bx[s * 4 + 1];
      o2 = bx[s * 4 + 2]; o3 = bx[s * 4 + 3];
    }
    rois[(size_t)tid * 4 + 0] = o0;
    rois[(size_t)tid * 4 + 1] = o1;
    rois[(size_t)tid * 4 + 2] = o2;
    rois[(size_t)tid * 4 + 3] = o3;
  }
}

// ---------------- host-side launcher ---------------------------------------
extern "C" void kernel_launch(void* const* d_in, const int* in_sizes, int n_in,
                              void* d_out, int out_size, void* d_ws, size_t ws_size,
                              hipStream_t stream) {
  const float* x    = (const float*)d_in[0];
  const float* cw3  = (const float*)d_in[1];
  const float* cb3  = (const float*)d_in[2];
  const float* clsw = (const float*)d_in[3];
  const float* clsb = (const float*)d_in[4];
  const float* regw = (const float*)d_in[5];
  const float* regb = (const float*)d_in[6];
  const int*   imh  = (const int*)d_in[7];
  const int*   imw  = (const int*)d_in[8];
  float* out = (float*)d_out;
  char*  ws  = (char*)d_ws;

  // workspace layout (bytes, all 64B aligned)
  bf16_t* xp  = (bf16_t*)(ws + 0);             // 32,169,984 : padded NHWC input bf16
  bf16_t* wT  = (bf16_t*)(ws + 32169984);      //  9,437,184 : [kk][n][c] weights bf16
  bf16_t* wc  = (bf16_t*)(ws + 41607168);      //     81,920 : combined 1x1 weights
  bf16_t* fm  = (bf16_t*)(ws + 41689088);      // 15,564,800 : conv feature map bf16
  float*  boxes  = (float*)(ws + 57253888);    //  2,188,800 : decoded boxes
  float*  scores = (float*)(ws + 59442688);    //    547,200 : objectness scores
  unsigned int* hist = (unsigned int*)(ws + 59989888);       // 16,384
  int* tinfo = (int*)(ws + 60006272);                        // 64
  unsigned long long* keys = (unsigned long long*)(ws + 60006336); // 65,536
  float* topBoxes = (float*)(ws + 60071872);                 // 96,000

  k_pad_input<<<62832, 256, 0, stream>>>(x, xp);                 // HP*WP*CIN / 256
  k_prep_w  <<<18432, 256, 0, stream>>>(cw3, wT);                // 9*512*1024 / 256
  k_prep_wc <<<160,   256, 0, stream>>>(clsw, regw, wc);
  k_conv3x3 <<<475,   256, 0, stream>>>(xp, wT, cb3, fm);        // 15200/32 tiles
  k_conv1x1 <<<594,   256, 0, stream>>>(fm, wc, clsb, regb, out);// ceil(4750/8)
  k_decode  <<<(NA + 255) / 256, 256, 0, stream>>>(out, boxes, scores, imh, imw);
  k_init    <<<32, 256, 0, stream>>>(hist, keys, tinfo);
  k_hist    <<<(NA + 255) / 256, 256, 0, stream>>>(scores, hist);
  k_thresh  <<<1, 32, 0, stream>>>(hist, tinfo);
  k_compact <<<(NA + 255) / 256, 256, 0, stream>>>(scores, keys, tinfo);
  k_sort    <<<1, 1024, SORT_N * sizeof(unsigned long long), stream>>>(keys);
  k_gather  <<<(PRE_N + 255) / 256, 256, 0, stream>>>(keys, boxes, topBoxes);
  k_nms     <<<1, 1024, PRE_N * 4 * sizeof(float), stream>>>(topBoxes, out + ROIS_OFF);
}
